// MultiHeadAttention_3350074491087
// MI455X (gfx1250) — matrix operand order repaired and checked by reading, not compiled, not run on hardware
//
#include <hip/hip_runtime.h>

// ---------------------------------------------------------------------------
// MultiHeadAttention (S=4096, d_model=1024, 16 heads, d_k=64, causal)
// bf16 WMMA (v_wmma_f32_16x16x32_bf16), f32 accumulation, async K staging.
// ---------------------------------------------------------------------------

#define SEQ     4096
#define DMODEL  1024
#define NHEADS  16
#define DK      64
#define FA_WAVES 4

typedef __bf16 bf16;
typedef __attribute__((ext_vector_type(16))) __bf16 bf16x16;
typedef __attribute__((ext_vector_type(8)))  __bf16 bf16x8;
typedef __attribute__((ext_vector_type(8)))  float  f32x8;

union ABFrag { bf16x16 v; bf16x8 h[2]; };

// ---- WMMA wrapper -----------------------------------------------------------
__device__ __forceinline__ f32x8 wmma_bf16(bf16x16 a, bf16x16 b, f32x8 c) {
  return __builtin_amdgcn_wmma_f32_16x16x32_bf16(
      /*neg_a=*/false, a, /*neg_b=*/false, b,
      /*c_mod=*/(short)0, c, /*reuse_a=*/false, /*reuse_b=*/false);
}

// ---- A fragment: 16x32 (MxK) bf16, row-major source, leading dim ld --------
__device__ __forceinline__ bf16x16 load_a_frag(const bf16* src, int ld) {
  const int lane = threadIdx.x & 31;
  const int m  = lane & 15;
  const int kb = (lane >> 4) << 3;   // 0 or 8
  ABFrag f;
  f.h[0] = *(const bf16x8*)(src + m * ld + kb);
  f.h[1] = *(const bf16x8*)(src + m * ld + kb + 16);
  return f.v;
}

// ---- B fragment: 32x16 (KxN) bf16, row-major source, leading dim ld --------
__device__ __forceinline__ bf16x16 load_b_frag(const bf16* src, int ld) {
  const int lane = threadIdx.x & 31;
  const int n  = lane & 15;
  const int kb = (lane >> 4) << 3;
  ABFrag f;
#pragma unroll
  for (int s = 0; s < 8; ++s) {
    f.v[s]     = src[(kb + s) * ld + n];
    f.v[8 + s] = src[(16 + kb + s) * ld + n];
  }
  return f.v;
}

// ---- CDNA5 async copy: 16B global -> LDS, per lane, tracked by ASYNCcnt ----
__device__ __forceinline__ void async_ld_b128(unsigned lds_byte_off, const bf16* g) {
  asm volatile("global_load_async_to_lds_b128 %0, %1, off"
               :: "v"(lds_byte_off), "v"(g) : "memory");
}

// ---- conversion kernels -----------------------------------------------------
__global__ void __launch_bounds__(256) k_cvt_bf16(const float* __restrict__ x,
                                                  bf16* __restrict__ xb, int n) {
  int i = blockIdx.x * blockDim.x + threadIdx.x;
  if (i < n) xb[i] = (bf16)x[i];
}

// wt[k][n] = w[n][k]
__global__ void __launch_bounds__(256) k_transpose_w(const float* __restrict__ w,
                                                     bf16* __restrict__ wt) {
  int i = blockIdx.x * blockDim.x + threadIdx.x;
  int n = i >> 10;
  int k = i & (DMODEL - 1);
  wt[k * DMODEL + n] = (bf16)w[i];
}

// ---- GEMM: Y = A[MxK] * Bt[KxN] + bias, per-wave 32x64 tile -----------------
// MODE 0: bf16 natural  MODE 1: bf16 transposed (Kt)  MODE 2: f32  MODE 3: bf16*0.125
template <int MODE>
__global__ void __launch_bounds__(256, 1)
k_gemm(const bf16* __restrict__ A, const bf16* __restrict__ Bt,
       const float* __restrict__ bias, void* __restrict__ out) {
  const int wid  = blockIdx.x * (blockDim.x >> 5) + (threadIdx.x >> 5);
  const int row0 = (wid >> 4) * 32;
  const int col0 = (wid & 15) * 64;

  const bf16* a_base = A + row0 * DMODEL;
  f32x8 acc[2][4] = {};
  for (int k = 0; k < DMODEL; k += 32) {
    __builtin_prefetch(a_base + k + 128, 0, 1);
    bf16x16 a0 = load_a_frag(a_base + k, DMODEL);
    bf16x16 a1 = load_a_frag(a_base + 16 * DMODEL + k, DMODEL);
#pragma unroll
    for (int t = 0; t < 4; ++t) {
      bf16x16 b = load_b_frag(Bt + k * DMODEL + col0 + t * 16, DMODEL);
      acc[0][t] = wmma_bf16(a0, b, acc[0][t]);
      acc[1][t] = wmma_bf16(a1, b, acc[1][t]);
    }
  }

  const int lane = threadIdx.x & 31;
  const int nloc = lane & 15;
  const int mh   = (lane >> 4) * 8;
#pragma unroll
  for (int half = 0; half < 2; ++half) {
#pragma unroll
    for (int t = 0; t < 4; ++t) {
      const int n  = col0 + t * 16 + nloc;
      const float bv = bias[n];
#pragma unroll
      for (int r = 0; r < 8; ++r) {
        const int m = row0 + half * 16 + r + mh;
        float v = acc[half][t][r] + bv;
        if (MODE == 3) v *= 0.125f;                       // 1/sqrt(64)
        if (MODE == 0 || MODE == 3) ((bf16*)out)[m * DMODEL + n] = (bf16)v;
        else if (MODE == 1)         ((bf16*)out)[n * SEQ + m]    = (bf16)v;
        else                        ((float*)out)[m * DMODEL + n] = v;
      }
    }
  }
}

// ---- flash attention helpers -----------------------------------------------
// Prefetch one 64-key K^T tile (dk rows 0..63 x keys key0..key0+63) to LDS.
// Layout in kst: [kc(2)][row(32)][key(64)] bf16; issues 16 async b128 copies.
__device__ __forceinline__ void prefetch_k(bf16* kst, const bf16* kt0,
                                           const bf16* kt1, int key0) {
  const int lane = threadIdx.x & 31;
  const bf16* g0 = kt0 + lane * SEQ + key0;
  const bf16* g1 = kt1 + lane * SEQ + key0;
  const unsigned l0 = (unsigned)(size_t)(kst + lane * 64);
  const unsigned l1 = (unsigned)(size_t)(kst + 2048 + lane * 64);
#pragma unroll
  for (int i = 0; i < 8; ++i) {
    async_ld_b128(l0 + i * 16, g0 + i * 8);
    async_ld_b128(l1 + i * 16, g1 + i * 8);
  }
}

// One 64-key flash block: scores from LDS-staged K, softmax, P x V.
template <bool MASKED>
__device__ __forceinline__ void fa_block(const bf16* kst, const bf16* vb,
                                         bf16* pslab, bf16x16 qa0, bf16x16 qa1,
                                         int key0, int q0,
                                         float (&mrow)[8], float (&lrow)[8],
                                         f32x8 (&acc)[4]) {
  const int lane = threadIdx.x & 31;
  const int half = lane >> 4;
  const int nloc = lane & 15;

  // ---- scores: four 16x16 key subtiles, K=64 chained WMMAs ----
  f32x8 s[4];
#pragma unroll
  for (int sub = 0; sub < 4; ++sub) {
    bf16x16 b0 = load_b_frag(kst + sub * 16, 64);          // dk 0..31
    bf16x16 b1 = load_b_frag(kst + 2048 + sub * 16, 64);   // dk 32..63
    f32x8 c = {};
    c = wmma_bf16(qa1, b1, c);
    c = wmma_bf16(qa0, b0, c);
    s[sub] = c;
  }

  if (MASKED) {
#pragma unroll
    for (int sub = 0; sub < 4; ++sub) {
      const int key = key0 + sub * 16 + nloc;
#pragma unroll
      for (int r = 0; r < 8; ++r) {
        const int qrow = q0 + r + half * 8;
        s[sub][r] = (key <= qrow) ? s[sub][r] : -INFINITY;
      }
    }
  }

  // ---- online softmax (row reductions across 16-lane halves) ----
#pragma unroll
  for (int r = 0; r < 8; ++r) {
    float v = fmaxf(fmaxf(s[0][r], s[1][r]), fmaxf(s[2][r], s[3][r]));
    v = fmaxf(v, __shfl_xor(v, 1, 32));
    v = fmaxf(v, __shfl_xor(v, 2, 32));
    v = fmaxf(v, __shfl_xor(v, 4, 32));
    v = fmaxf(v, __shfl_xor(v, 8, 32));
    const float mnew = fmaxf(mrow[r], v);

    float rs = 0.f;
#pragma unroll
    for (int sub = 0; sub < 4; ++sub) {
      const float p = __expf(s[sub][r] - mnew);
      s[sub][r] = p;
      rs += p;
    }
    rs += __shfl_xor(rs, 1, 32);
    rs += __shfl_xor(rs, 2, 32);
    rs += __shfl_xor(rs, 4, 32);
    rs += __shfl_xor(rs, 8, 32);

    const float alpha = __expf(mrow[r] - mnew);
    lrow[r] = lrow[r] * alpha + rs;
    mrow[r] = mnew;
#pragma unroll
    for (int t = 0; t < 4; ++t) acc[t][r] *= alpha;
  }

  // ---- P: C/D layout -> A layout via per-wave LDS slab (16x64) ----
#pragma unroll
  for (int sub = 0; sub < 4; ++sub)
#pragma unroll
    for (int r = 0; r < 8; ++r)
      pslab[(r + half * 8) * 64 + sub * 16 + nloc] = (bf16)s[sub][r];
  bf16x16 pa0 = load_a_frag(pslab, 64);        // keys key0+0..31
  bf16x16 pa1 = load_a_frag(pslab + 32, 64);   // keys key0+32..63

  // ---- ctx += P (16x64) x V (64x64), V direct from global ----
#pragma unroll
  for (int t = 0; t < 4; ++t) {
    bf16x16 bv0 = load_b_frag(vb + key0 * DMODEL + t * 16, DMODEL);
    acc[t] = wmma_bf16(pa0, bv0, acc[t]);
    bf16x16 bv1 = load_b_frag(vb + (key0 + 32) * DMODEL + t * 16, DMODEL);
    acc[t] = wmma_bf16(pa1, bv1, acc[t]);
  }
}

// ---- Flash attention core: 1 wave per (head, 16-query tile) -----------------
// Qb: bf16 [SEQ][DMODEL] (pre-scaled by 1/8)   Kt: bf16 [DMODEL][SEQ]
// Vb: bf16 [SEQ][DMODEL]                        Ctx: bf16 [SEQ][DMODEL]
__global__ void __launch_bounds__(32 * FA_WAVES, 1)
k_flash_attn(const bf16* __restrict__ Qb, const bf16* __restrict__ Kt,
             const bf16* __restrict__ Vb, bf16* __restrict__ Ctx) {
  // per wave: 2 x (2x32x64) K staging (16 KB) + 16x64 P slab (2 KB)
  __shared__ __attribute__((aligned(16))) bf16 kstage[FA_WAVES * 2 * 4096];
  __shared__ __attribute__((aligned(16))) bf16 pstage[FA_WAVES * 16 * 64];

  const int wslot = threadIdx.x >> 5;
  const int wid   = blockIdx.x * FA_WAVES + wslot;
  const int h     = wid >> 8;      // SEQ/16 = 256 q-tiles per head
  const int q0    = (wid & 255) * 16;
  const int lane  = threadIdx.x & 31;
  const int half  = lane >> 4;
  const int nloc  = lane & 15;

  const bf16* qsrc = Qb + q0 * DMODEL + h * DK;
  bf16x16 qa0 = load_a_frag(qsrc, DMODEL);
  bf16x16 qa1 = load_a_frag(qsrc + 32, DMODEL);

  const bf16* kt0 = Kt + (h * DK + 0)  * SEQ;
  const bf16* kt1 = Kt + (h * DK + 32) * SEQ;
  const bf16* vb  = Vb + h * DK;
  bf16* kbuf0 = kstage + wslot * 8192;
  bf16* kbuf1 = kbuf0 + 4096;
  bf16* pslab = pstage + wslot * (16 * 64);

  float mrow[8], lrow[8];
  f32x8 acc[4] = {};
#pragma unroll
  for (int r = 0; r < 8; ++r) { mrow[r] = -INFINITY; lrow[r] = 0.f; }

  // causal: nfull fully-unmasked 64-key blocks + exactly one masked block
  const int nfull = q0 >> 6;

  prefetch_k(kbuf0, kt0, kt1, 0);
  for (int kb = 0; kb < nfull; ++kb) {
    prefetch_k((kb & 1) ? kbuf0 : kbuf1, kt0, kt1, (kb + 1) * 64);
    asm volatile("s_wait_asynccnt 0x10" ::: "memory");   // older 16 done, next 16 in flight
    fa_block<false>((kb & 1) ? kbuf1 : kbuf0, vb, pslab, qa0, qa1,
                    kb * 64, q0, mrow, lrow, acc);
  }
  asm volatile("s_wait_asynccnt 0x0" ::: "memory");
  fa_block<true>((nfull & 1) ? kbuf1 : kbuf0, vb, pslab, qa0, qa1,
                 nfull * 64, q0, mrow, lrow, acc);

  // ---- normalize, store ctx tile ----
#pragma unroll
  for (int r = 0; r < 8; ++r) {
    const float inv = 1.0f / lrow[r];
    const int m = q0 + r + half * 8;
#pragma unroll
    for (int t = 0; t < 4; ++t)
      Ctx[m * DMODEL + h * DK + t * 16 + nloc] = (bf16)(acc[t][r] * inv);
  }
}

// ---------------------------------------------------------------------------
extern "C" void kernel_launch(void* const* d_in, const int* in_sizes, int n_in,
                              void* d_out, int out_size, void* d_ws, size_t ws_size,
                              hipStream_t stream) {
  const float* x  = (const float*)d_in[0];
  const float* wq = (const float*)d_in[1];
  const float* bq = (const float*)d_in[2];
  const float* wk = (const float*)d_in[3];
  const float* bk = (const float*)d_in[4];
  const float* wv = (const float*)d_in[5];
  const float* bv = (const float*)d_in[6];
  const float* wo = (const float*)d_in[7];
  const float* bo = (const float*)d_in[8];

  char* ws = (char*)d_ws;
  const size_t MB = 1ull << 20;
  bf16* Xb   = (bf16*)(ws + 0 * MB);
  bf16* Wqt  = (bf16*)(ws + 8 * MB);
  bf16* Wkt  = (bf16*)(ws + 10 * MB);
  bf16* Wvt  = (bf16*)(ws + 12 * MB);
  bf16* Wot  = (bf16*)(ws + 14 * MB);
  bf16* Qb   = (bf16*)(ws + 16 * MB);   // pre-scaled by 1/8
  bf16* Ktr  = (bf16*)(ws + 24 * MB);   // per-head transposed K
  bf16* Vb   = (bf16*)(ws + 32 * MB);
  bf16* Ctxb = (bf16*)(ws + 40 * MB);

  // 1) conversions
  k_cvt_bf16<<<(SEQ * DMODEL) / 256, 256, 0, stream>>>(x, Xb, SEQ * DMODEL);
  k_transpose_w<<<(DMODEL * DMODEL) / 256, 256, 0, stream>>>(wq, Wqt);
  k_transpose_w<<<(DMODEL * DMODEL) / 256, 256, 0, stream>>>(wk, Wkt);
  k_transpose_w<<<(DMODEL * DMODEL) / 256, 256, 0, stream>>>(wv, Wvt);
  k_transpose_w<<<(DMODEL * DMODEL) / 256, 256, 0, stream>>>(wo, Wot);

  // 2) projections
  const int gemm_blocks = (SEQ / 32) * (DMODEL / 64) / 8;  // 256
  k_gemm<3><<<gemm_blocks, 256, 0, stream>>>(Xb, Wqt, bq, Qb);
  k_gemm<1><<<gemm_blocks, 256, 0, stream>>>(Xb, Wkt, bk, Ktr);
  k_gemm<0><<<gemm_blocks, 256, 0, stream>>>(Xb, Wvt, bv, Vb);

  // 3) flash attention: 4096 wave tiles, FA_WAVES waves/block
  k_flash_attn<<<NHEADS * (SEQ / 16) / FA_WAVES, 32 * FA_WAVES, 0, stream>>>(
      Qb, Ktr, Vb, Ctxb);

  // 4) output projection -> fp32 d_out
  k_gemm<2><<<gemm_blocks, 256, 0, stream>>>(Ctxb, Wot, bo, (float*)d_out);
}
